// SchNet_48163763257859
// MI455X (gfx1250) — compile-verified, matrix operand order checked by reading, not run on hardware
//
#include <hip/hip_runtime.h>
#include <hip/hip_bf16.h>
#include <math.h>

// ---------------------------------------------------------------------------
// SchNet forward, MI455X (gfx1250, wave32, WMMA bf16 16x16x32, fp32 accum).
// - Weights pre-swizzled to per-lane WMMA B-fragment order (32B vector LDS
//   reads, zero packing VALU).
// - Dual n-tile accumulator interleave to hide WMMA->WMMA RAW hazard NOPs.
// - Block-uniform bounds fast paths (no per-element exec-mask churn).
// ---------------------------------------------------------------------------

typedef __attribute__((ext_vector_type(16))) __bf16          v16bf;
typedef __attribute__((ext_vector_type(8)))  float           v8f;
typedef __attribute__((ext_vector_type(16))) unsigned short  v16us;
typedef __attribute__((ext_vector_type(8)))  unsigned short  v8us;
typedef __attribute__((ext_vector_type(4)))  unsigned short  v4us;

namespace {
constexpr int   N_ATOM = 50000;
constexpr int   E_EDGE = 600000;
constexpr int   H      = 128;
constexpr int   G      = 50;
constexpr int   GP     = 64;     // G padded to WMMA-friendly K
constexpr int   L      = 6;
constexpr int   NC     = 500;
constexpr int   NM     = 100;
constexpr float CUTOFF = 10.0f;
constexpr float STEP   = CUTOFF / (float)(G - 1);
constexpr float COEFF  = -0.5f / (STEP * STEP);
constexpr float LOG2F_ = 0.69314718055994531f;
}

#define DEVFN __device__ __forceinline__

DEVFN unsigned short f2bf(float f) {
  union { float f; unsigned u; } v; v.f = f;
  unsigned r = v.u + 0x7FFFu + ((v.u >> 16) & 1u);   // round-to-nearest-even
  return (unsigned short)(r >> 16);
}
DEVFN float sspf(float x) {   // shifted softplus
  float sp = (x > 20.0f) ? x : __logf(1.0f + __expf(x));
  return sp - LOG2F_;
}
DEVFN v8f wmma_bf16(v16bf a, v16bf b, v8f c) {
#if defined(__HIP_DEVICE_COMPILE__)
  return __builtin_amdgcn_wmma_f32_16x16x32_bf16(false, a, false, b, (short)0, c, false, false);
#else
  return c;
#endif
}

// A fragment from row-major LDS [rows x stride] -> two aligned 16B loads.
DEVFN v16bf load_frag_a(const unsigned short* s, int stride, int m0, int k0, int lane) {
  const int m = m0 + (lane & 15);
  const int hi = (lane >> 4) * 8;
  union { struct { v8us lo, hi; } p; v16bf b; } t;
  t.p.lo = *(const v8us*)(s + m * stride + k0 + hi);
  t.p.hi = *(const v8us*)(s + m * stride + k0 + 16 + hi);
  return t.b;
}
// B fragment from swizzled LDS: frag (kb,nt) stored [32 lanes][16 elems]
// contiguous -> one aligned 32B load per lane.
DEVFN v16bf load_frag_b_sw(const unsigned short* s, int kb, int nt, int lane) {
  union { v16us u; v16bf b; } t;
  t.u = *(const v16us*)(s + (((kb << 3) + nt) << 9) + (lane << 4));
  return t.b;
}

// ---------------------------------------------------------------------------
// Weight swizzle: src [mats][Ksrc][128] f32 -> dst [mats][Kpad/32][8][32][16]
// bf16 in per-lane fragment order (zero-padded for k >= Ksrc).
// ---------------------------------------------------------------------------
__global__ void swz_weight_kernel(const float* __restrict__ src,
                                  unsigned short* __restrict__ dst,
                                  int Ksrc, int Kpad, int mats) {
  int i = blockIdx.x * 256 + threadIdx.x;      // over mats*Kpad*128
  int per = Kpad * H;
  if (i >= mats * per) return;
  int m = i / per, r = i % per;
  int fi = r >> 9;                              // fragment (512 elems)
  int rr = r & 511;
  int lane = rr >> 4, e = rr & 15;
  int kb = fi >> 3, nt = fi & 7;
  int hi = (lane >> 4) * 8;
  int k = kb * 32 + ((e < 8) ? (hi + e) : (16 + hi + (e - 8)));
  int n = nt * 16 + (lane & 15);
  float v = (k < Ksrc) ? src[((size_t)m * Ksrc + k) * H + n] : 0.0f;
  dst[i] = f2bf(v);
}

__global__ void zero_f32_kernel(float* __restrict__ p, int n) {
  int i = blockIdx.x * 256 + threadIdx.x;
  if (i < n) p[i] = 0.0f;
}

__global__ void embed_kernel(const int* __restrict__ z,
                             const float* __restrict__ emb,
                             float* __restrict__ h) {
  int i = blockIdx.x * 256 + threadIdx.x;      // over N*H
  int n = i >> 7, f = i & 127;
  h[i] = emb[z[n] * H + f];
}

__global__ void edge_geom_kernel(const float* __restrict__ pos,
                                 const int* __restrict__ er,
                                 const int* __restrict__ ec,
                                 float* __restrict__ dist,
                                 float* __restrict__ Cc) {
  int e = blockIdx.x * 256 + threadIdx.x;
  if (e >= E_EDGE) return;
  int j = er[e], i = ec[e];
  float dx = pos[j * 3 + 0] - pos[i * 3 + 0];
  float dy = pos[j * 3 + 1] - pos[i * 3 + 1];
  float dz = pos[j * 3 + 2] - pos[i * 3 + 2];
  float d = sqrtf(dx * dx + dy * dy + dz * dz);
  dist[e] = d;
  Cc[e] = 0.5f * (__cosf(d * (float)(M_PI) / CUTOFF) + 1.0f);
}

// Stage 64 rows of f32 [*,128] into LDS bf16 (float4 vectorized).
DEVFN void stage_in64(unsigned short* sIn, const float* in, int r0, int nrows, int t) {
  if (r0 + 64 <= nrows) {                       // fast path: full tile
    for (int i = t; i < 64 * H / 4; i += 256) {
      int row = i >> 5, q = i & 31;
      float4 v = ((const float4*)in)[(size_t)(r0 + row) * (H / 4) + q];
      v4us o; o.x = f2bf(v.x); o.y = f2bf(v.y); o.z = f2bf(v.z); o.w = f2bf(v.w);
      *(v4us*)(sIn + i * 4) = o;
    }
  } else {
    for (int i = t; i < 64 * H / 4; i += 256) {
      int row = i >> 5, q = i & 31;
      int g = r0 + row;
      v4us o;
      if (g < nrows) {
        float4 v = ((const float4*)in)[(size_t)g * (H / 4) + q];
        o.x = f2bf(v.x); o.y = f2bf(v.y); o.z = f2bf(v.z); o.w = f2bf(v.w);
      } else {
        o = (v4us)0;
      }
      *(v4us*)(sIn + i * 4) = o;
    }
  }
}
// Stage swizzled weights into LDS (16B chunks).
DEVFN void stage_w(unsigned short* sW, const unsigned short* w, int nus, int t) {
  const int nq = nus / 8;
  for (int i = t; i < nq; i += 256)
    ((uint4*)sW)[i] = ((const uint4*)w)[i];
}

// Dual n-tile K-loop: interleaves two accumulator chains to hide WMMA RAW NOPs.
DEVFN void gemm_pair(const unsigned short* sA, const unsigned short* sW,
                     int mt, int nt0, int nt1, int lane, int nkb,
                     v8f& acc0, v8f& acc1) {
  acc0 = (v8f){}; acc1 = (v8f){};
  for (int kk = 0; kk < nkb; ++kk) {
    v16bf a  = load_frag_a(sA, (nkb == 2) ? GP : H, mt * 16, kk * 32, lane);
    v16bf b0 = load_frag_b_sw(sW, kk, nt0, lane);
    v16bf b1 = load_frag_b_sw(sW, kk, nt1, lane);
    acc0 = wmma_bf16(a, b0, acc0);
    acc1 = wmma_bf16(a, b1, acc1);
  }
}

// ---------------------------------------------------------------------------
// xf = h @ W (N x 128 x 128, no bias). 64 rows / block, 8 waves.
// ---------------------------------------------------------------------------
__global__ __launch_bounds__(256) void xf_gemm_kernel(
    const float* __restrict__ in, const unsigned short* __restrict__ wgt,
    float* __restrict__ out, int nrows) {
  __shared__ __attribute__((aligned(32))) unsigned short sIn[64 * H];  // 16 KB
  __shared__ __attribute__((aligned(32))) unsigned short sW[H * H];    // 32 KB
  const int t = threadIdx.x, lane = t & 31, w = t >> 5;
  const int r0 = blockIdx.x * 64;
  stage_in64(sIn, in, r0, nrows, t);
  stage_w(sW, wgt, H * H, t);
  __syncthreads();
  const int mt = w >> 1, nbase = (w & 1) * 4;
  const int hi8 = (lane >> 4) * 8, nlane = lane & 15;
  const bool full = (r0 + 64 <= nrows);
#pragma unroll
  for (int np = 0; np < 2; ++np) {
    int nt0 = nbase + np * 2, nt1 = nt0 + 1;
    v8f acc0, acc1;
    gemm_pair(sIn, sW, mt, nt0, nt1, lane, 4, acc0, acc1);
#pragma unroll
    for (int half = 0; half < 2; ++half) {
      v8f& acc = half ? acc1 : acc0;
      int col = (half ? nt1 : nt0) * 16 + nlane;
      int row = r0 + mt * 16 + hi8;
      float* op = out + (size_t)row * H + col;
      if (full) {
#pragma unroll
        for (int r = 0; r < 8; ++r) op[r * H] = acc[r];
      } else {
#pragma unroll
        for (int r = 0; r < 8; ++r)
          if (row + r < nrows) op[r * H] = acc[r];
      }
    }
  }
}

// ---------------------------------------------------------------------------
// out = ssp(in @ W1 + b1) @ W2 + b2 (+ res). 64 rows / block, sW reused.
// ---------------------------------------------------------------------------
__global__ __launch_bounds__(256) void node_fused_kernel(
    const float* __restrict__ in, const unsigned short* __restrict__ w1,
    const float* __restrict__ b1, const unsigned short* __restrict__ w2,
    const float* __restrict__ b2, float* __restrict__ out,
    const float* __restrict__ res, int nrows) {
  __shared__ __attribute__((aligned(32))) unsigned short sIn[64 * H];   // 16 KB
  __shared__ __attribute__((aligned(32))) unsigned short sW[H * H];     // 32 KB
  __shared__ __attribute__((aligned(32))) unsigned short sMid[64 * H];  // 16 KB
  const int t = threadIdx.x, lane = t & 31, w = t >> 5;
  const int r0 = blockIdx.x * 64;
  stage_in64(sIn, in, r0, nrows, t);
  stage_w(sW, w1, H * H, t);
  __syncthreads();
  const int mt = w >> 1, nbase = (w & 1) * 4;
  const int hi8 = (lane >> 4) * 8, nlane = lane & 15;
  const bool full = (r0 + 64 <= nrows);
#pragma unroll
  for (int np = 0; np < 2; ++np) {
    int nt0 = nbase + np * 2, nt1 = nt0 + 1;
    v8f acc0, acc1;
    gemm_pair(sIn, sW, mt, nt0, nt1, lane, 4, acc0, acc1);
#pragma unroll
    for (int half = 0; half < 2; ++half) {
      v8f& acc = half ? acc1 : acc0;
      int col = (half ? nt1 : nt0) * 16 + nlane;
      float bias = b1[col];
#pragma unroll
      for (int r = 0; r < 8; ++r)
        sMid[(mt * 16 + hi8 + r) * H + col] = f2bf(sspf(acc[r] + bias));
    }
  }
  __syncthreads();
  stage_w(sW, w2, H * H, t);
  __syncthreads();
#pragma unroll
  for (int np = 0; np < 2; ++np) {
    int nt0 = nbase + np * 2, nt1 = nt0 + 1;
    v8f acc0, acc1;
    gemm_pair(sMid, sW, mt, nt0, nt1, lane, 4, acc0, acc1);
#pragma unroll
    for (int half = 0; half < 2; ++half) {
      v8f& acc = half ? acc1 : acc0;
      int col = (half ? nt1 : nt0) * 16 + nlane;
      float bias = b2[col];
      int row = r0 + mt * 16 + hi8;
      float* op = out + (size_t)row * H + col;
      const float* rp = res ? (res + (size_t)row * H + col) : nullptr;
      if (full) {
        if (rp) {
#pragma unroll
          for (int r = 0; r < 8; ++r) op[r * H] = acc[r] + bias + rp[r * H];
        } else {
#pragma unroll
          for (int r = 0; r < 8; ++r) op[r * H] = acc[r] + bias;
        }
      } else {
#pragma unroll
        for (int r = 0; r < 8; ++r)
          if (row + r < nrows) {
            float v = acc[r] + bias;
            if (rp) v += rp[r * H];
            op[r * H] = v;
          }
      }
    }
  }
}

// ---------------------------------------------------------------------------
// Fused edge pipeline, 64 edges / block:
//   A = exp(COEFF*(d-off)^2) built in LDS from d (edge_attr never in HBM)
//   T = ssp(A @ W1 + b1); W = (T @ W2 + b2) * C
//   msg = xf[edge_row] * W ; atomicAdd(agg[edge_col], msg)
// ---------------------------------------------------------------------------
__global__ __launch_bounds__(256) void edge_layer_kernel(
    const float* __restrict__ dist, const float* __restrict__ Cc,
    const int* __restrict__ erow, const int* __restrict__ ecol,
    const unsigned short* __restrict__ w1, const float* __restrict__ b1,
    const unsigned short* __restrict__ w2, const float* __restrict__ b2,
    const float* __restrict__ xf, float* __restrict__ agg) {
  __shared__ __attribute__((aligned(32))) unsigned short sA[64 * GP];   // 8 KB
  __shared__ __attribute__((aligned(32))) unsigned short sW[H * H];     // 32 KB
  __shared__ __attribute__((aligned(32))) unsigned short sT[64 * H];    // 16 KB
  __shared__ int   sER[64];
  __shared__ int   sEC[64];
  __shared__ float sC[64];
  const int t = threadIdx.x, lane = t & 31, w = t >> 5;
  const int e0 = blockIdx.x * 64;
  if (t < 64) {
    sER[t] = erow[e0 + t];
    sEC[t] = ecol[e0 + t];
    sC[t]  = Cc[e0 + t];
  }
  {  // Gaussian smearing: thread t -> row t>>2, 16 gaussians, vector store
    const int m = t >> 2;
    const float dv = dist[e0 + m];
    const int gbase = (t & 3) * 16;
    v16us o;
#pragma unroll
    for (int j = 0; j < 16; ++j) {
      int g = gbase + j;
      float v = 0.0f;
      if (g < G) {
        float x = dv - (float)g * STEP;
        v = __expf(COEFF * x * x);
      }
      o[j] = f2bf(v);
    }
    *(v16us*)(sA + m * GP + gbase) = o;
  }
  stage_w(sW, w1, GP * H, t);
  __syncthreads();
  const int mt = w >> 1, nbase = (w & 1) * 4;
  const int hi8 = (lane >> 4) * 8, nlane = lane & 15;
#pragma unroll
  for (int np = 0; np < 2; ++np) {   // GEMM1: [64 x 64] @ [64 x 128]
    int nt0 = nbase + np * 2, nt1 = nt0 + 1;
    v8f acc0, acc1;
    gemm_pair(sA, sW, mt, nt0, nt1, lane, 2, acc0, acc1);
#pragma unroll
    for (int half = 0; half < 2; ++half) {
      v8f& acc = half ? acc1 : acc0;
      int col = (half ? nt1 : nt0) * 16 + nlane;
      float bias = b1[col];
#pragma unroll
      for (int r = 0; r < 8; ++r)
        sT[(mt * 16 + hi8 + r) * H + col] = f2bf(sspf(acc[r] + bias));
    }
  }
  __syncthreads();
  stage_w(sW, w2, H * H, t);
  __syncthreads();
#pragma unroll
  for (int np = 0; np < 2; ++np) {   // GEMM2 + cutoff + gather + scatter-add
    int nt0 = nbase + np * 2, nt1 = nt0 + 1;
    v8f acc0, acc1;
    gemm_pair(sT, sW, mt, nt0, nt1, lane, 4, acc0, acc1);
#pragma unroll
    for (int half = 0; half < 2; ++half) {
      v8f& acc = half ? acc1 : acc0;
      int col = (half ? nt1 : nt0) * 16 + nlane;
      float bias = b2[col];
#pragma unroll
      for (int r = 0; r < 8; ++r) {
        int le = mt * 16 + hi8 + r;
        float wv = (acc[r] + bias) * sC[le];
        float msg = xf[(size_t)sER[le] * H + col] * wv;
        atomicAdd(&agg[(size_t)sEC[le] * H + col], msg);
      }
    }
  }
}

// ---------------------------------------------------------------------------
// Pooling + head
// ---------------------------------------------------------------------------
__global__ void seg_sum_kernel(const float* __restrict__ src,
                               const int* __restrict__ idx,
                               float* __restrict__ dst, int nrows) {
  int i = blockIdx.x * 256 + threadIdx.x;
  if (i >= nrows * H) return;
  int n = i >> 7, f = i & 127;
  atomicAdd(&dst[(size_t)idx[n] * H + f], src[i]);
}

__global__ void head_kernel(const float* __restrict__ mol,
                            const float* __restrict__ w1,
                            const float* __restrict__ b1,
                            const float* __restrict__ w2,
                            const float* __restrict__ b2,
                            float* __restrict__ out) {
  __shared__ float s[64];
  const int m = blockIdx.x, j = threadIdx.x;   // 64 threads
  float acc = b1[j];
  for (int k = 0; k < H; ++k) acc += mol[m * H + k] * w1[k * 64 + j];
  s[j] = sspf(acc);
  __syncthreads();
  if (j == 0) {
    float o = b2[0];
    for (int jj = 0; jj < 64; ++jj) o += s[jj] * w2[jj];
    out[m] = o;
  }
}

// ---------------------------------------------------------------------------
// Host orchestration
// ---------------------------------------------------------------------------
extern "C" void kernel_launch(void* const* d_in, const int* in_sizes, int n_in,
                              void* d_out, int out_size, void* d_ws, size_t ws_size,
                              hipStream_t stream) {
  (void)in_sizes; (void)n_in; (void)out_size; (void)ws_size;
  const int*   z       = (const int*)  d_in[0];
  const float* pos     = (const float*)d_in[1];
  const int*   erow    = (const int*)  d_in[2];
  const int*   ecol    = (const int*)  d_in[3];
  const int*   a2c     = (const int*)  d_in[4];
  const int*   c2m     = (const int*)  d_in[5];
  const float* emb     = (const float*)d_in[6];
  const float* mlp_w1  = (const float*)d_in[7];
  const float* mlp_b1  = (const float*)d_in[8];
  const float* mlp_w2  = (const float*)d_in[9];
  const float* mlp_b2  = (const float*)d_in[10];
  const float* c1w     = (const float*)d_in[11];
  const float* c2w     = (const float*)d_in[12];
  const float* c2b     = (const float*)d_in[13];
  const float* ilw     = (const float*)d_in[14];
  const float* ilb     = (const float*)d_in[15];
  const float* l1w     = (const float*)d_in[16];
  const float* l1b     = (const float*)d_in[17];
  const float* l2w     = (const float*)d_in[18];
  const float* l2b     = (const float*)d_in[19];
  const float* hw1     = (const float*)d_in[20];
  const float* hb1     = (const float*)d_in[21];
  const float* hw2     = (const float*)d_in[22];
  const float* hb2     = (const float*)d_in[23];
  float* out = (float*)d_out;

  char* p = (char*)d_ws;
  auto carve = [&](size_t bytes) -> void* {
    void* r = (void*)p; p += (bytes + 255) & ~(size_t)255; return r;
  };
  float* wd   = (float*)carve((size_t)E_EDGE * 4);
  float* wC   = (float*)carve((size_t)E_EDGE * 4);
  float* wh   = (float*)carve((size_t)N_ATOM * H * 4);
  float* wxf  = (float*)carve((size_t)N_ATOM * H * 4);
  float* wagg = (float*)carve((size_t)N_ATOM * H * 4);
  float* wh2  = (float*)carve((size_t)N_ATOM * H * 4);
  float* wcnf = (float*)carve((size_t)NC * H * 4);
  float* wmol = (float*)carve((size_t)NM * H * 4);
  unsigned short* w1p   = (unsigned short*)carve((size_t)L * GP * H * 2);
  unsigned short* w2b   = (unsigned short*)carve((size_t)L * H * H * 2);
  unsigned short* c1b16 = (unsigned short*)carve((size_t)L * H * H * 2);
  unsigned short* c2b16 = (unsigned short*)carve((size_t)L * H * H * 2);
  unsigned short* il16  = (unsigned short*)carve((size_t)L * H * H * 2);
  unsigned short* l1b16 = (unsigned short*)carve((size_t)H * H * 2);
  unsigned short* l2b16 = (unsigned short*)carve((size_t)H * H * 2);

  const int LHH = L * H * H;
  // Weight prep: convert + pad + swizzle into WMMA B-fragment order
  swz_weight_kernel<<<(L * GP * H + 255) / 256, 256, 0, stream>>>(mlp_w1, w1p, G, GP, L);
  swz_weight_kernel<<<(LHH + 255) / 256, 256, 0, stream>>>(mlp_w2, w2b, H, H, L);
  swz_weight_kernel<<<(LHH + 255) / 256, 256, 0, stream>>>(c1w, c1b16, H, H, L);
  swz_weight_kernel<<<(LHH + 255) / 256, 256, 0, stream>>>(c2w, c2b16, H, H, L);
  swz_weight_kernel<<<(LHH + 255) / 256, 256, 0, stream>>>(ilw, il16, H, H, L);
  swz_weight_kernel<<<(H * H + 255) / 256, 256, 0, stream>>>(l1w, l1b16, H, H, 1);
  swz_weight_kernel<<<(H * H + 255) / 256, 256, 0, stream>>>(l2w, l2b16, H, H, 1);

  // Embedding + edge geometry
  embed_kernel<<<(N_ATOM * H) / 256, 256, 0, stream>>>(z, emb, wh);
  edge_geom_kernel<<<(E_EDGE + 255) / 256, 256, 0, stream>>>(pos, erow, ecol, wd, wC);

  const int node_blocks = (N_ATOM + 63) / 64;   // 782
  const int edge_blocks = E_EDGE / 64;          // 9375 (E divisible by 64)

  for (int l = 0; l < L; ++l) {
    zero_f32_kernel<<<(N_ATOM * H) / 256, 256, 0, stream>>>(wagg, N_ATOM * H);
    xf_gemm_kernel<<<node_blocks, 256, 0, stream>>>(
        wh, c1b16 + (size_t)l * H * H, wxf, N_ATOM);
    edge_layer_kernel<<<edge_blocks, 256, 0, stream>>>(
        wd, wC, erow, ecol,
        w1p + (size_t)l * GP * H, mlp_b1 + (size_t)l * H,
        w2b + (size_t)l * H * H,  mlp_b2 + (size_t)l * H,
        wxf, wagg);
    node_fused_kernel<<<node_blocks, 256, 0, stream>>>(
        wagg,
        c2b16 + (size_t)l * H * H, c2b + (size_t)l * H,
        il16  + (size_t)l * H * H, ilb + (size_t)l * H,
        wh, wh, N_ATOM);                         // residual: h = h + x
  }

  // Final node MLP: h2 = ssp(h@lin1+b1)@lin2+b2
  node_fused_kernel<<<node_blocks, 256, 0, stream>>>(
      wh, l1b16, l1b, l2b16, l2b, wh2, nullptr, N_ATOM);

  // Pooling
  zero_f32_kernel<<<(NC * H + 255) / 256, 256, 0, stream>>>(wcnf, NC * H);
  zero_f32_kernel<<<(NM * H + 255) / 256, 256, 0, stream>>>(wmol, NM * H);
  seg_sum_kernel<<<(N_ATOM * H + 255) / 256, 256, 0, stream>>>(wh2, a2c, wcnf, N_ATOM);
  seg_sum_kernel<<<(NC * H + 255) / 256, 256, 0, stream>>>(wcnf, c2m, wmol, NC);

  // Head
  head_kernel<<<NM, 64, 0, stream>>>(wmol, hw1, hb1, hw2, hb2, out);
}
